// RNNModelScratch_18829136625664
// MI455X (gfx1250) — matrix operand order repaired and checked by reading, not compiled, not run on hardware
//
#include <hip/hip_runtime.h>
#include <hip/hip_bf16.h>

// LSTM LM: gather(4 gates) -> 128-step recurrence -> [4096x32000] projection.
// V=32000, HD=512, B=32, T=128.
// Precision: bf16 WMMA (v_wmma_f32_16x16x32_bf16) with f32 accumulation.
// Roofline: projection writes 524MB f32 (>=22us @23.3TB/s) and is ~134 GFLOP;
// bf16 WMMA keeps compute ~balanced with HBM. Recurrence (8.6 GFLOP) is
// latency bound -> 32 persistent WGs + per-step grid barrier, weights in L2.
// Projection is register-blocked 2Mx2N per wave: 4 WMMAs per 4 fragment loads.

#define V_  32000
#define HD_ 512
#define B_  32
#define T_  128
#define VH  32512  // V_+HD_

typedef __attribute__((ext_vector_type(16))) __bf16 v16bf;
typedef __attribute__((ext_vector_type(8)))  float  v8f;

// ---- WMMA fragment packing (CDNA5 ISA 7.12.2, wave32) --------------------
// A 16x32 bf16: lanes 0-15 -> M=lane, VGPR r<4: K=2r,2r+1 ; r>=4: K=16+2(r-4);
// lanes 16-31 -> same M, K offset +8.
__device__ __forceinline__ v16bf a_frag(const __bf16* base, int mbase, int k0, int lane) {
  const __bf16* row = base + (mbase + (lane & 15)) * HD_;
  int khi = (lane & 16) ? 8 : 0;
  v16bf a;
#pragma unroll
  for (int r = 0; r < 8; ++r) {
    int k = k0 + ((r & 4) << 2) + khi + ((r & 3) << 1);
    a[2*r]   = row[k];
    a[2*r+1] = row[k+1];
  }
  return a;
}

// B 32x16 bf16 (K x N): lanes 0-15 -> N=lane, VGPR r<4: K=2r,2r+1 ; r>=4: K=16+2(r-4);
// lanes 16-31 -> N=lane-16, K offset +8.  wn: column-major, wn[n*HD_ + k].
__device__ __forceinline__ v16bf b_frag(const __bf16* wn, int k0, int lane) {
  const __bf16* col = wn + (lane & 15) * HD_;
  int kb = k0 + ((lane & 16) ? 8 : 0);
  v16bf b;
#pragma unroll
  for (int r = 0; r < 8; ++r) {
    int k = kb + ((r & 4) << 2) + ((r & 3) << 1);
    b[2*r]   = col[k];
    b[2*r+1] = col[k+1];
  }
  return b;
}

__device__ __forceinline__ float sigmoidf_(float x) { return 1.f / (1.f + __expf(-x)); }

// ---- kernel 0: reset grid-barrier counter (ws is poisoned between calls) --
__global__ void init_kernel(unsigned* cnt) { if (threadIdx.x == 0) *cnt = 0u; }

// ---- kernel 1: embedding gather + bias  E[t][g][b][j] = Wg_x[j, X[b,t]] + bg[j]
__global__ void __launch_bounds__(256) gather_kernel(
    const int* __restrict__ X,
    const float* __restrict__ Wf, const float* __restrict__ bf,
    const float* __restrict__ Wi, const float* __restrict__ bi,
    const float* __restrict__ Wo, const float* __restrict__ bo,
    const float* __restrict__ Wc, const float* __restrict__ bc,
    float* __restrict__ E) {
  size_t idx = (size_t)blockIdx.x * 256 + threadIdx.x;   // [T][4][B][HD], j fastest
  int j = (int)(idx & (HD_ - 1));
  int b = (int)((idx >> 9) & (B_ - 1));
  int g = (int)((idx >> 14) & 3);
  int t = (int)(idx >> 16);
  int x = X[b * T_ + t];
  const float* W; const float* bb;
  switch (g) { case 0: W = Wf; bb = bf; break; case 1: W = Wi; bb = bi; break;
               case 2: W = Wo; bb = bo; break; default: W = Wc; bb = bc; }
  E[idx] = W[(size_t)j * VH + x] + bb[j];
}

// ---- kernel 2: hidden-part gate weights f32 -> bf16, column-major per output j
// WhB[((g*HD + j)*HD) + k] = Wg[j, V_ + k]
__global__ void __launch_bounds__(256) cvt_wh_kernel(
    const float* __restrict__ Wf, const float* __restrict__ Wi,
    const float* __restrict__ Wo, const float* __restrict__ Wc,
    __bf16* __restrict__ WhB) {
  size_t idx = (size_t)blockIdx.x * 256 + threadIdx.x;   // 4*512*512
  int k = (int)(idx & (HD_ - 1));
  int j = (int)((idx >> 9) & (HD_ - 1));
  int g = (int)(idx >> 18);
  const float* W = (g == 0) ? Wf : (g == 1) ? Wi : (g == 2) ? Wo : Wc;
  WhB[idx] = (__bf16)W[(size_t)j * VH + V_ + k];
}

// ---- kernel 3: Wy f32 -> bf16 (rows of Wy are columns of Wy^T: already K-contig)
__global__ void __launch_bounds__(256) cvt_wy_kernel(const float* __restrict__ Wy,
                                                     __bf16* __restrict__ WyB) {
  size_t idx = (size_t)blockIdx.x * 256 + threadIdx.x;   // 32000*512
  WyB[idx] = (__bf16)Wy[idx];
}

// ---- kernel 4: persistent LSTM recurrence -------------------------------
// 32 workgroups x 256 threads (8 waves). WG w owns hidden slice j in
// [16w,16w+16) for all 4 gates (so c/h updates are local). Wave (g,mtile)
// computes a 16x16 preactivation tile via 16 bf16 WMMAs (K=512).
// h is broadcast via a double-buffered global bf16 buffer + grid barrier.
__global__ void __launch_bounds__(256) lstm_kernel(
    const float* __restrict__ H0, const float* __restrict__ C0,
    const float* __restrict__ E,  const __bf16* __restrict__ WhB,
    __bf16* __restrict__ hbuf, __bf16* __restrict__ Hs,
    float* __restrict__ Hout, float* __restrict__ Cout,
    unsigned* __restrict__ cnt) {
  __shared__ __bf16 hA[B_ * HD_];     // 32 KB: full h, A operand (M=32, K=512)
  __shared__ float  G[4 * B_ * 16];   //  8 KB: gate preactivations for this slice
  __shared__ float  cS[B_ * 16];      //  2 KB: cell state slice

  const int tid   = threadIdx.x;
  const int lane  = tid & 31;
  const int wave  = tid >> 5;
  const int g     = wave >> 1;        // gate 0..3
  const int mtile = wave & 1;         // batch half
  const int jbase = blockIdx.x * 16;  // owned hidden columns
  const unsigned nwg = gridDim.x;

  for (int e = tid; e < B_ * 16; e += 256)
    cS[e] = C0[(e >> 4) * HD_ + jbase + (e & 15)];

  const __bf16* Wn = WhB + ((size_t)g * HD_ + jbase) * HD_;  // 16 cols, K-contig

  for (int t = 0; t < T_; ++t) {
    // stage h (all 32x512) into LDS
    if (t == 0) {
      for (int e = tid; e < B_ * HD_; e += 256) hA[e] = (__bf16)H0[e];
    } else {
      const uint4* src = (const uint4*)(hbuf + ((t + 1) & 1) * (B_ * HD_));
      uint4* dst = (uint4*)hA;
      for (int e = tid; e < B_ * HD_ / 8; e += 256) dst[e] = src[e];
    }
    __syncthreads();

    v8f acc = {};
#pragma unroll 4
    for (int kb = 0; kb < HD_ / 32; ++kb) {
      v16bf a = a_frag(hA, mtile * 16, kb * 32, lane);
      v16bf b = b_frag(Wn, kb * 32, lane);
      acc = __builtin_amdgcn_wmma_f32_16x16x32_bf16(false, a, false, b,
                                                    (short)0, acc, false, false);
    }
    // add gathered embedding term, park preactivations in LDS
    {
      int j = lane & 15;
      int rb = mtile * 16 + ((lane & 16) ? 8 : 0);
      const float* Eg = E + ((size_t)(t * 4 + g) * B_) * HD_;
#pragma unroll
      for (int r = 0; r < 8; ++r) {
        int row = rb + r;
        G[(g * B_ + row) * 16 + j] = acc[r] + Eg[(size_t)row * HD_ + jbase + j];
      }
    }
    __syncthreads();

    // pointwise gates: 512 slice elements, 2 per thread
    __bf16* hnext = hbuf + (t & 1) * (B_ * HD_);
    for (int e = tid; e < B_ * 16; e += 256) {
      int b = e >> 4, j = e & 15;
      float f  = sigmoidf_(G[(0 * B_ + b) * 16 + j]);
      float i  = sigmoidf_(G[(1 * B_ + b) * 16 + j]);
      float o  = sigmoidf_(G[(2 * B_ + b) * 16 + j]);
      float ct = tanhf(G[(3 * B_ + b) * 16 + j]);
      float c  = f * cS[e] + i * ct;
      float h  = o * tanhf(c);
      cS[e] = c;
      hnext[b * HD_ + jbase + j] = (__bf16)h;
      Hs[((size_t)t * B_ + b) * HD_ + jbase + j] = (__bf16)h;
      if (t == T_ - 1) {
        Hout[b * HD_ + jbase + j] = h;
        Cout[b * HD_ + jbase + j] = c;
      }
    }
    // grid barrier (monotonic target: (t+1)*nwg; counter reset by init_kernel)
    __syncthreads();
    if (tid == 0) {
      __threadfence();
      atomicAdd(cnt, 1u);
      unsigned target = (unsigned)(t + 1) * nwg;
      while (atomicAdd(cnt, 0u) < target) __builtin_amdgcn_s_sleep(2);
      __threadfence();
    }
    __syncthreads();
  }
}

// ---- kernel 5: output projection Y[4096,32000] = Hs(bf16) @ Wy^T + by ----
// block = 8 waves, tile = 32 rows x 256 cols. Each wave register-blocks a
// 32x32 output (2M x 2N tiles): 4 WMMAs per 2 A-frag (LDS) + 2 B-frag (L2)
// loads, halving per-FLOP fragment traffic. WyB (32MB) is L2-resident.
__global__ void __launch_bounds__(256) proj_kernel(
    const __bf16* __restrict__ Hs, const __bf16* __restrict__ WyB,
    const float* __restrict__ by, float* __restrict__ Y) {
  __shared__ __bf16 Atile[32 * HD_];  // 32 KB
  const int tid = threadIdx.x;
  const int bid = blockIdx.x;
  const int ng  = bid % 125;          // 125 col-groups of 256
  const int mg  = bid / 125;          // 128 row-groups of 32
  {
    const uint4* src = (const uint4*)(Hs + (size_t)mg * 32 * HD_);
    uint4* dst = (uint4*)Atile;
    for (int e = tid; e < 32 * HD_ / 8; e += 256) dst[e] = src[e];
  }
  __syncthreads();
  const int lane = tid & 31;
  const int wave = tid >> 5;
  const int nb0  = ng * 256 + wave * 32;        // two adjacent 16-col tiles
  const __bf16* Wn0 = WyB + (size_t)nb0 * HD_;
  const __bf16* Wn1 = Wn0 + 16 * HD_;
  v8f acc00 = {}, acc01 = {}, acc10 = {}, acc11 = {};
#pragma unroll 2
  for (int kb = 0; kb < HD_ / 32; ++kb) {
    v16bf a0 = a_frag(Atile, 0,  kb * 32, lane);
    v16bf a1 = a_frag(Atile, 16, kb * 32, lane);
    v16bf b0 = b_frag(Wn0, kb * 32, lane);
    v16bf b1 = b_frag(Wn1, kb * 32, lane);
    acc00 = __builtin_amdgcn_wmma_f32_16x16x32_bf16(false, a0, false, b0,
                                                    (short)0, acc00, false, false);
    acc01 = __builtin_amdgcn_wmma_f32_16x16x32_bf16(false, a0, false, b1,
                                                    (short)0, acc01, false, false);
    acc10 = __builtin_amdgcn_wmma_f32_16x16x32_bf16(false, a1, false, b0,
                                                    (short)0, acc10, false, false);
    acc11 = __builtin_amdgcn_wmma_f32_16x16x32_bf16(false, a1, false, b1,
                                                    (short)0, acc11, false, false);
  }
  const int j     = lane & 15;
  const int rlo   = (lane & 16) ? 8 : 0;
  const int n0    = nb0 + j;
  const int n1    = nb0 + 16 + j;
  const float bias0 = by[n0];
  const float bias1 = by[n1];
  const int r0    = mg * 32 + rlo;        // rows for mtile 0
#pragma unroll
  for (int r = 0; r < 8; ++r) {
    Y[(size_t)(r0 + r)      * V_ + n0] = acc00[r] + bias0;
    Y[(size_t)(r0 + r)      * V_ + n1] = acc01[r] + bias1;
    Y[(size_t)(r0 + 16 + r) * V_ + n0] = acc10[r] + bias0;
    Y[(size_t)(r0 + 16 + r) * V_ + n1] = acc11[r] + bias1;
  }
}

extern "C" void kernel_launch(void* const* d_in, const int* in_sizes, int n_in,
                              void* d_out, int out_size, void* d_ws, size_t ws_size,
                              hipStream_t stream) {
  (void)in_sizes; (void)n_in; (void)out_size; (void)ws_size;
  const int*   X  = (const int*)  d_in[0];
  const float* H0 = (const float*)d_in[1];
  const float* C0 = (const float*)d_in[2];
  const float* Wf = (const float*)d_in[3];
  const float* bf = (const float*)d_in[4];
  const float* Wi = (const float*)d_in[5];
  const float* bi = (const float*)d_in[6];
  const float* Wo = (const float*)d_in[7];
  const float* bo = (const float*)d_in[8];
  const float* Wc = (const float*)d_in[9];
  const float* bc = (const float*)d_in[10];
  const float* Wy = (const float*)d_in[11];
  const float* by = (const float*)d_in[12];

  // workspace carve-out (~70 MB total)
  char* ws = (char*)d_ws;
  float*    E    = (float*)ws;   ws += (size_t)T_ * 4 * B_ * HD_ * sizeof(float);  // 32 MiB
  __bf16*   Hs   = (__bf16*)ws;  ws += (size_t)T_ * B_ * HD_ * 2;                  //  4 MiB
  __bf16*   WyB  = (__bf16*)ws;  ws += (size_t)V_ * HD_ * 2;                       // ~31 MiB
  __bf16*   WhB  = (__bf16*)ws;  ws += (size_t)4 * HD_ * HD_ * 2;                  //  2 MiB
  __bf16*   hbuf = (__bf16*)ws;  ws += (size_t)2 * B_ * HD_ * 2;                   // 64 KiB
  unsigned* cnt  = (unsigned*)ws;

  float* Y    = (float*)d_out;
  float* Hout = Y + (size_t)T_ * B_ * V_;   // tuple: Y, then H, then C
  float* Cout = Hout + (size_t)B_ * HD_;

  init_kernel<<<1, 64, 0, stream>>>(cnt);
  gather_kernel<<<(T_ * 4 * B_ * HD_) / 256, 256, 0, stream>>>(
      X, Wf, bf, Wi, bi, Wo, bo, Wc, bc, E);
  cvt_wh_kernel<<<(4 * HD_ * HD_) / 256, 256, 0, stream>>>(Wf, Wi, Wo, Wc, WhB);
  cvt_wy_kernel<<<(V_ * HD_) / 256, 256, 0, stream>>>(Wy, WyB);
  lstm_kernel<<<HD_ / 16, 256, 0, stream>>>(H0, C0, E, WhB, hbuf, Hs, Hout, Cout, cnt);
  proj_kernel<<<(T_ * B_ / 32) * (V_ / 256), 256, 0, stream>>>(Hs, WyB, by, Y);
}
